// DKGMBlock_58574763983532
// MI455X (gfx1250) — matrix operand order repaired and checked by reading, not compiled, not run on hardware
//
#include <hip/hip_runtime.h>
#include <hip/hip_bf16.h>
#include <math.h>

#define IMG   512
#define CELL  16
#define NCELL 32
#define ORI   9
#define PLANE (IMG * IMG)

typedef __attribute__((ext_vector_type(16))) _Float16 v16h;
typedef __attribute__((ext_vector_type(8)))  float    v8f;

// ---------------------------------------------------------------------------
// Workspace layout:
//   [0, EMB_BYTES)            : emb as _Float16, (16,3,512,512)  -> 25,165,824 B
//   [EMB_BYTES, +GRAY_BYTES)  : gray as float,   (16,512,512)    -> 16,777,216 B
// ---------------------------------------------------------------------------
#define EMB_ELEMS  (16 * 3 * PLANE)
#define EMB_BYTES  (EMB_ELEMS * 2)
#define GRAY_ELEMS (16 * PLANE)

// ===========================================================================
// Kernel A: per-pixel HLS (H, L) + grayscale.  Memory-bound elementwise pass.
// Writes emb channels 0 (H) and 1 (L) as f16, gray as f32.
// ===========================================================================
__global__ __launch_bounds__(256) void hls_gray_kernel(
    const float* __restrict__ x, _Float16* __restrict__ emb,
    float* __restrict__ gray)
{
    int idx = blockIdx.x * blockDim.x + threadIdx.x;          // pixel id
    if (idx >= 16 * PLANE) return;
    int img = idx / PLANE;
    int pix = idx - img * PLANE;

    const float* base = x + (size_t)img * 3 * PLANE + pix;
    float r = base[0 * PLANE];
    float g = base[1 * PLANE];
    float b = base[2 * PLANE];

    float vmax = fmaxf(fmaxf(r, g), b);
    float vmin = fminf(fminf(r, g), b);
    float diff = vmax - vmin;
    float l    = (vmax + vmin) * 0.5f;
    bool  nz   = diff > 0.0f;
    float safe = nz ? diff : 1.0f;

    float h = (vmax == r) ? 60.0f * (g - b) / safe
            : (vmax == g) ? 120.0f + 60.0f * (b - r) / safe
                          : 240.0f + 60.0f * (r - g) / safe;
    if (h < 0.0f) h += 360.0f;
    if (!nz) h = 0.0f;

    _Float16* eb = emb + (size_t)img * 3 * PLANE + pix;
    eb[0 * PLANE] = (_Float16)h;
    eb[1 * PLANE] = (_Float16)l;

    gray[(size_t)img * PLANE + pix] = 0.299f * r + 0.587f * g + 0.114f * b;
}

// ===========================================================================
// Kernel B: HOG per cell.  One 256-thread block per 16x16 cell.
//   - per-pixel gradients + sqrt/atan2 + 9-bin LDS histogram (ds_add_f32)
//   - 9 threads trace Bresenham lines (exact Python-int() semantics) into an
//     LDS 16x16 tile (line offsets stay inside the cell -> no global atomics)
//   - clip(.,0,10)/10 -> emb channel 2 (f16)
// ===========================================================================
__global__ __launch_bounds__(256) void hog_cell_kernel(
    const float* __restrict__ gray, _Float16* __restrict__ emb)
{
    __shared__ float hist[ORI];
    __shared__ float tile[CELL * CELL];

    int bid = blockIdx.x;                 // img * 1024 + cy * 32 + cx
    int img = bid >> 10;
    int cy  = (bid >> 5) & 31;
    int cx  = bid & 31;

    int t  = threadIdx.x;
    int ly = t >> 4;
    int lx = t & 15;
    int y  = cy * CELL + ly;
    int xx = cx * CELL + lx;

    if (t < ORI) hist[t] = 0.0f;
    tile[t] = 0.0f;
    __syncthreads();

    const float* gp = gray + (size_t)img * PLANE;
    float gr = (y >= 1 && y <= IMG - 2)
                 ? gp[(y + 1) * IMG + xx] - gp[(y - 1) * IMG + xx] : 0.0f;
    float gc = (xx >= 1 && xx <= IMG - 2)
                 ? gp[y * IMG + xx + 1] - gp[y * IMG + xx - 1] : 0.0f;
    float sq  = gr * gr + gc * gc;
    float mag = (sq > 0.0f) ? sqrtf(sq) : 0.0f;
    float ang = atan2f(gr, gc) * (180.0f / 3.14159265358979323846f);
    ang = fmodf(ang, 180.0f);
    if (ang < 0.0f) ang += 180.0f;
    int bin = (int)(ang * (ORI / 180.0f));
    bin = min(max(bin, 0), ORI - 1);
    atomicAdd(&hist[bin], mag);
    __syncthreads();

    if (t < ORI) {
        float val = hist[t] * (1.0f / (CELL * CELL));
        float mid = 3.14159265358979323846f * (t + 0.5f) / ORI;
        float drf = 7.0f * sinf(mid);
        float dcf = 7.0f * cosf(mid);
        int r0 = (int)(8.0f - dcf);   // C cast truncates toward zero == int()
        int c0 = (int)(8.0f + drf);
        int r1 = (int)(8.0f + dcf);
        int c1 = (int)(8.0f - drf);

        int r = r0, c = c0;
        int adr = abs(r1 - r0), adc = abs(c1 - c0);
        int sc = (c1 - c0 > 0) ? 1 : -1;
        int sr = (r1 - r0 > 0) ? 1 : -1;
        bool steep = adr > adc;
        if (steep) {
            int tmp = r;  r  = c;   c  = tmp;
            tmp = adr;  adr = adc; adc = tmp;
            tmp = sr;   sr  = sc;  sc  = tmp;
        }
        int d = 2 * adr - adc;
        for (int i = 0; i < adc; ++i) {
            int pr = steep ? c : r;
            int pc = steep ? r : c;
            atomicAdd(&tile[pr * CELL + pc], val);
            while (d >= 0) { r += sr; d -= 2 * adc; }
            c += sc; d += 2 * adr;
        }
        atomicAdd(&tile[r1 * CELL + c1], val);
    }
    __syncthreads();

    float hog = tile[t];
    hog = fminf(fmaxf(hog, 0.0f), 10.0f) * 0.1f;
    emb[((size_t)img * 3 + 2) * PLANE + (size_t)y * IMG + xx] = (_Float16)hog;
}

// ===========================================================================
// Kernel C: 3x3 conv (3->3 ch) via V_WMMA_F32_16X16X32_F16 + residual add.
// One wave handles a 16-pixel strip of one row:
//   A (16x32 f16): M = pixel in strip, K = ci*9 + tap (27 valid, rest 0)
//   B (32x16 f16): N = out channel (3 valid), K as above
//   D (16x16 f32): lane n<3 holds 8 consecutive pixels -> coalesced stores
// ===========================================================================
__global__ __launch_bounds__(256) void conv_wmma_kernel(
    const _Float16* __restrict__ emb, const float* __restrict__ xin,
    const float* __restrict__ convw, const float* __restrict__ convb,
    float* __restrict__ out)
{
    int wave = (blockIdx.x * blockDim.x + threadIdx.x) >> 5;
    // 16 imgs * 512 rows * 32 strips
    int img  = wave >> 14;
    int rem  = wave & 16383;
    int y    = rem >> 5;
    int x0   = (rem & 31) << 4;

    int lane = threadIdx.x & 31;
    int half = lane & 16;                 // 0 or 16

    // ---- Build A operand: lane (lane&15) = pixel M; halves split K ----
    v16h a;
    {
        int m  = lane & 15;
        int xp = x0 + m;
        int kb = half ? 8 : 0;
        size_t b3 = (size_t)img * 3;
#pragma unroll
        for (int e = 0; e < 16; ++e) {
            int v = e >> 1;
            int K = kb + ((v & 4) ? 16 : 0) + ((v & 3) << 1) + (e & 1);
            _Float16 val = (_Float16)0.0f;
            if (K < 27) {
                int ci  = K / 9;
                int tap = K - ci * 9;
                int ky  = tap / 3 - 1;
                int kx  = tap - (tap / 3) * 3 - 1;
                int yy  = y + ky;
                int xg  = xp + kx;
                if (yy >= 0 && yy < IMG && xg >= 0 && xg < IMG)
                    val = emb[(b3 + ci) * PLANE + (size_t)yy * IMG + xg];
            }
            a[e] = val;
        }
    }

    // ---- Build B operand: lane&15 = out channel N; K striped per ISA ----
    v16h bm;
    {
        int n   = lane & 15;
        int kb2 = half ? 8 : 0;
#pragma unroll
        for (int e = 0; e < 16; ++e) {
            int K = kb2 + ((e & 8) ? e + 8 : e);
            float w = (n < 3 && K < 27) ? convw[n * 27 + K] : 0.0f;
            bm[e] = (_Float16)w;
        }
    }

    v8f c = {};
    c = __builtin_amdgcn_wmma_f32_16x16x32_f16(
            /*neg_a=*/false, a, /*neg_b=*/false, bm,
            /*c_mod=*/(short)0, c, /*reuse_a=*/false, /*reuse_b=*/false);

    // ---- Residual + bias + store: lane n<3 owns 8 consecutive pixels ----
    int n = lane & 15;
    if (n < 3) {
        int xbase = x0 + (half ? 8 : 0);
        size_t off = ((size_t)img * 3 + n) * PLANE + (size_t)y * IMG + xbase;
        const float* xs = xin + off;
        float*       op = out + off;
        float bias = convb[n];
#pragma unroll
        for (int j = 0; j < 8; ++j)
            op[j] = xs[j] + c[j] + bias;
    }
}

// ===========================================================================
extern "C" void kernel_launch(void* const* d_in, const int* in_sizes, int n_in,
                              void* d_out, int out_size, void* d_ws, size_t ws_size,
                              hipStream_t stream) {
    const float* x     = (const float*)d_in[0];   // (16,3,512,512) f32
    const float* convw = (const float*)d_in[1];   // (3,3,3,3) OIHW f32
    const float* convb = (const float*)d_in[2];   // (3,) f32
    float* out = (float*)d_out;

    _Float16* emb = (_Float16*)d_ws;
    float* gray   = (float*)((char*)d_ws + EMB_BYTES);

    // Kernel A: 16*512*512 pixels, 256 thr/blk
    {
        int total  = 16 * PLANE;
        int blocks = (total + 255) / 256;
        hls_gray_kernel<<<blocks, 256, 0, stream>>>(x, emb, gray);
    }
    // Kernel B: one block per cell: 16 imgs * 32 * 32
    {
        hog_cell_kernel<<<16 * NCELL * NCELL, 256, 0, stream>>>(gray, emb);
    }
    // Kernel C: one wave per 16-pixel strip: 16*512*32 waves, 8 waves/blk
    {
        int waves  = 16 * IMG * (IMG / 16);
        int blocks = waves / 8;
        conv_wmma_kernel<<<blocks, 256, 0, stream>>>(emb, x, convw, convb, out);
    }
}
